// Qwen3VLVisionAttention_69922067578960
// MI455X (gfx1250) — matrix operand order repaired and checked by reading, not compiled, not run on hardware
//
#include <hip/hip_runtime.h>
#include <hip/hip_bf16.h>

typedef _Float16 f16;
typedef _Float16 v16h __attribute__((ext_vector_type(16)));
typedef float    v8f  __attribute__((ext_vector_type(8)));

union FragH {
  v16h v;
  f16  h[16];
  uint4 q[2];
};

__device__ __forceinline__ v8f wmma16(const FragH& a, const FragH& b, v8f c) {
  // D = A(16x32 f16) * B(32x16 f16) + C(16x16 f32)
  return __builtin_amdgcn_wmma_f32_16x16x32_f16(false, a.v, false, b.v,
                                                (short)0, c, false, false);
}

// B-fragment: two 8-f16 runs at base and base+16 (N = lane%16 fixed, table K order)
__device__ __forceinline__ void load_bfrag(const f16* base, FragH& f) {
  f.q[0] = *(const uint4*)(base);
  f.q[1] = *(const uint4*)(base + 16);
}

#define TOTAL   8192
#define DMODEL  1280
#define HEADS   16
#define HD      80
#define HDP     96          // head_dim padded to 3*32 for clean WMMA K-steps
#define CHUNKS  8
#define CLEN    1024
#define SCALE   0.11180339887498948f   // 1/sqrt(80)

// ---------------------------------------------------------------------------
// Kernel 0: f32 [K x N] row-major  ->  f16 [N x K] row-major (N-major weights)
// ---------------------------------------------------------------------------
__global__ void transpose_cvt(const float* __restrict__ in, f16* __restrict__ out,
                              int K, int N) {
  size_t idx = (size_t)blockIdx.x * blockDim.x + threadIdx.x;
  size_t total = (size_t)K * N;
  if (idx >= total) return;
  size_t n = idx / K, k = idx % K;
  out[idx] = (f16)in[k * (size_t)N + n];
}

// ---------------------------------------------------------------------------
// A-operand staging for f32 sources: raw float4 loads (issued early, as one
// clause), converted to the f16 A-fragment layout only at the point of use.
// ---------------------------------------------------------------------------
struct ARaw { float4 t0, t1, t2, t3; };

__device__ __forceinline__ void load_araw(const float* __restrict__ X, int row,
                                          int k0, int hf, ARaw& a) {
  const float4* x4 = (const float4*)(X + (size_t)row * DMODEL + k0);
  a.t0 = x4[2 * hf];     a.t1 = x4[2 * hf + 1];
  a.t2 = x4[4 + 2 * hf]; a.t3 = x4[5 + 2 * hf];
}

__device__ __forceinline__ FragH cvt_a(const ARaw& a) {
  FragH f;
  f.h[0]  = (f16)a.t0.x; f.h[1]  = (f16)a.t0.y; f.h[2]  = (f16)a.t0.z; f.h[3]  = (f16)a.t0.w;
  f.h[4]  = (f16)a.t1.x; f.h[5]  = (f16)a.t1.y; f.h[6]  = (f16)a.t1.z; f.h[7]  = (f16)a.t1.w;
  f.h[8]  = (f16)a.t2.x; f.h[9]  = (f16)a.t2.y; f.h[10] = (f16)a.t2.z; f.h[11] = (f16)a.t2.w;
  f.h[12] = (f16)a.t3.x; f.h[13] = (f16)a.t3.y; f.h[14] = (f16)a.t3.z; f.h[15] = (f16)a.t3.w;
  return f;
}

// ---------------------------------------------------------------------------
// Kernel 1: QKV GEMM (16 rows x 80 cols per WG == one head of q|k|v) + bias,
// then RoPE for q/k through LDS; writes:
//   Qr, Kr : f16 [C][H][L][96]  (d 80..95 zero-padded)
//   Vt     : f16 [C][H][80][L]  (d-major so PV B-fragments are contiguous)
// Inner loop is software-pipelined (double-buffered, static buffers).
// ---------------------------------------------------------------------------
__global__ void __launch_bounds__(32)
qkv_rope_kernel(const float* __restrict__ X, const f16* __restrict__ WqkvT,
                const float* __restrict__ bqkv, const float* __restrict__ cosp,
                const float* __restrict__ sinp, f16* __restrict__ Qr,
                f16* __restrict__ Kr, f16* __restrict__ Vt) {
  const int rt = blockIdx.x;          // row tile (16 rows of T)
  const int gy = blockIdx.y;          // 0..15 q, 16..31 k, 32..47 v heads
  const int lane = threadIdx.x;
  const int lm = lane & 15, hf = lane >> 4;
  const int colbase = gy * HD;

  v8f acc[5] = {};
  const int row = rt * 16 + lm;

  ARaw ar0, ar1;
  FragH fb0[5], fb1[5];

  load_araw(X, row, 0, hf, ar0);
#pragma unroll
  for (int nt = 0; nt < 5; ++nt)
    load_bfrag(WqkvT + (size_t)(colbase + nt * 16 + lm) * DMODEL + 8 * hf, fb0[nt]);

  for (int k0 = 0; k0 < DMODEL; k0 += 64) {
    // stage k0+32 while computing k0
    load_araw(X, row, k0 + 32, hf, ar1);
#pragma unroll
    for (int nt = 0; nt < 5; ++nt)
      load_bfrag(WqkvT + (size_t)(colbase + nt * 16 + lm) * DMODEL + (k0 + 32) + 8 * hf,
                 fb1[nt]);
    {
      FragH fa = cvt_a(ar0);
#pragma unroll
      for (int nt = 0; nt < 5; ++nt) acc[nt] = wmma16(fa, fb0[nt], acc[nt]);
    }
    // stage k0+64 while computing k0+32
    if (k0 + 64 < DMODEL) {
      load_araw(X, row, k0 + 64, hf, ar0);
#pragma unroll
      for (int nt = 0; nt < 5; ++nt)
        load_bfrag(WqkvT + (size_t)(colbase + nt * 16 + lm) * DMODEL + (k0 + 64) + 8 * hf,
                   fb0[nt]);
    }
    {
      FragH fa = cvt_a(ar1);
#pragma unroll
      for (int nt = 0; nt < 5; ++nt) acc[nt] = wmma16(fa, fb1[nt], acc[nt]);
    }
  }

  // ---- epilogue: bias, stage tile in LDS, RoPE, store f16 ----
  __shared__ float sOut[16][HD];
#pragma unroll
  for (int nt = 0; nt < 5; ++nt)
#pragma unroll
    for (int v = 0; v < 8; ++v)
      sOut[v + 8 * hf][nt * 16 + lm] = acc[nt][v] + bqkv[colbase + nt * 16 + lm];
  __syncthreads();

  const int tbase = rt * 16;
  for (int it = lane; it < 16 * HD; it += 32) {
    int r = it / HD, d = it % HD;
    int t = tbase + r;
    float val = sOut[r][d];
    float res = val;
    if (gy < 32) {  // rope for q and k
      float par = sOut[r][(d + 40) % HD];
      float rh = (d < 40) ? -par : par;
      res = val * cosp[(size_t)t * HD + d] + rh * sinp[(size_t)t * HD + d];
    }
    int c = t / CLEN, l = t % CLEN;
    if (gy < 16) {
      Qr[(((size_t)c * HEADS + gy) * CLEN + l) * HDP + d] = (f16)res;
    } else if (gy < 32) {
      Kr[(((size_t)c * HEADS + (gy - 16)) * CLEN + l) * HDP + d] = (f16)res;
    } else {
      Vt[(((size_t)c * HEADS + (gy - 32)) * HD + d) * CLEN + l] = (f16)res;
    }
  }
  if (gy < 32) {  // zero pad d = 80..95 for q/k
    f16* dst = (gy < 16) ? Qr : Kr;
    int hh = (gy < 16) ? gy : gy - 16;
    for (int it = lane; it < 16 * 16; it += 32) {
      int r = it / 16, d = HD + (it % 16);
      int t = tbase + r, c = t / CLEN, l = t % CLEN;
      dst[(((size_t)c * HEADS + hh) * CLEN + l) * HDP + d] = (f16)0.f;
    }
  }
}

// ---------------------------------------------------------------------------
// Kernel 2: block-diagonal flash attention, one (chunk, head, 16-row tile)
// per workgroup. All K/V fragments of an m-block are issued as one clause
// before the WMMAs; next block's K/V rows are prefetched during softmax.
// ---------------------------------------------------------------------------
__global__ void __launch_bounds__(32)
attn_kernel(const f16* __restrict__ Qr, const f16* __restrict__ Kr,
            const f16* __restrict__ Vt, f16* __restrict__ attnH) {
  const int l0 = blockIdx.x * 16;
  const int h  = blockIdx.y;
  const int c  = blockIdx.z;
  const int lane = threadIdx.x;
  const int lm = lane & 15, hf = lane >> 4;

  const size_t chb = ((size_t)c * HEADS + h);
  const f16* Qp = Qr + chb * CLEN * HDP;
  const f16* Kp = Kr + chb * CLEN * HDP;
  const f16* Vp = Vt + chb * HD * CLEN;

  // Q A-fragments for the 3 K-steps (d = 0..95)
  FragH qa[3];
  const f16* qrow = Qp + (size_t)(l0 + lm) * HDP;
#pragma unroll
  for (int ks = 0; ks < 3; ++ks) {
    qa[ks].q[0] = *(const uint4*)(qrow + ks * 32 + 8 * hf);
    qa[ks].q[1] = *(const uint4*)(qrow + ks * 32 + 16 + 8 * hf);
  }

  float rm[8], rl[8];
#pragma unroll
  for (int v = 0; v < 8; ++v) { rm[v] = -1e30f; rl[v] = 0.f; }
  v8f o[5] = {};

  __shared__ f16 sP[16][32];

  for (int m0 = 0; m0 < CLEN; m0 += 32) {
    // ---- issue ALL fragment loads for this m-block as one clause ----
    FragH kb[6], vb[5];
#pragma unroll
    for (int ks = 0; ks < 3; ++ks) {
      load_bfrag(Kp + (size_t)(m0 + lm) * HDP + ks * 32 + 8 * hf, kb[2 * ks]);
      load_bfrag(Kp + (size_t)(m0 + 16 + lm) * HDP + ks * 32 + 8 * hf, kb[2 * ks + 1]);
    }
#pragma unroll
    for (int nt = 0; nt < 5; ++nt)
      load_bfrag(Vp + (size_t)(nt * 16 + lm) * CLEN + m0 + 8 * hf, vb[nt]);

    // ---- QK^T: 6 WMMAs (V loads keep flying underneath) ----
    v8f s0 = {}, s1 = {};
#pragma unroll
    for (int ks = 0; ks < 3; ++ks) {
      s0 = wmma16(qa[ks], kb[2 * ks], s0);
      s1 = wmma16(qa[ks], kb[2 * ks + 1], s1);
    }

    // ---- warm the WGP cache for the next m-block (global_prefetch_b8) ----
    if (m0 + 32 < CLEN) {
      const f16* kn = Kp + (size_t)(m0 + 32 + lane) * HDP;   // 32 K rows, 192B each
      __builtin_prefetch(kn, 0, 0);
      __builtin_prefetch(kn + 64, 0, 0);
#pragma unroll
      for (int j = 0; j < 3; ++j) {                          // 80 V channel rows
        int r = lane + 32 * j;
        if (r < HD) __builtin_prefetch(Vp + (size_t)r * CLEN + m0 + 32, 0, 0);
      }
    }

    // ---- online softmax; stats replicated across each 16-lane half ----
    float alpha[8];
#pragma unroll
    for (int v = 0; v < 8; ++v) {
      float a = s0[v] * SCALE, b = s1[v] * SCALE;
      float mx = fmaxf(a, b);
#pragma unroll
      for (int m = 1; m <= 8; m <<= 1) mx = fmaxf(mx, __shfl_xor(mx, m, 32));
      float nm = fmaxf(rm[v], mx);
      float p0 = __expf(a - nm), p1 = __expf(b - nm);
      float sum = p0 + p1;
#pragma unroll
      for (int m = 1; m <= 8; m <<= 1) sum += __shfl_xor(sum, m, 32);
      float al = __expf(rm[v] - nm);
      rl[v] = rl[v] * al + sum;
      rm[v] = nm;
      alpha[v] = al;
      sP[v + 8 * hf][lm]      = (f16)p0;   // P in C-layout -> LDS
      sP[v + 8 * hf][16 + lm] = (f16)p1;
    }
#pragma unroll
    for (int nt = 0; nt < 5; ++nt)
#pragma unroll
      for (int v = 0; v < 8; ++v) o[nt][v] *= alpha[v];
    __syncthreads();

    // re-read P as A-fragment (layout redistribution via ds_load_b128)
    FragH pa;
    pa.q[0] = *(const uint4*)(&sP[lm][8 * hf]);
    pa.q[1] = *(const uint4*)(&sP[lm][16 + 8 * hf]);

#pragma unroll
    for (int nt = 0; nt < 5; ++nt) o[nt] = wmma16(pa, vb[nt], o[nt]);
    __syncthreads();
  }

  // ---- normalize and store attn output f16 [T][D] ----
  float inv[8];
#pragma unroll
  for (int v = 0; v < 8; ++v) inv[v] = 1.0f / rl[v];
#pragma unroll
  for (int nt = 0; nt < 5; ++nt)
#pragma unroll
    for (int v = 0; v < 8; ++v) {
      int t = c * CLEN + l0 + v + 8 * hf;
      int col = h * HD + nt * 16 + lm;
      attnH[(size_t)t * DMODEL + col] = (f16)(o[nt][v] * inv[v]);
    }
}

// ---------------------------------------------------------------------------
// Kernel 3: output projection, f16 operands, f32 accumulate + bias, f32 out.
// Software-pipelined like kernel 1 (A is already f16 here).
// ---------------------------------------------------------------------------
__global__ void __launch_bounds__(32)
proj_kernel(const f16* __restrict__ A, const f16* __restrict__ WprojT,
            const float* __restrict__ bproj, float* __restrict__ out) {
  const int rt = blockIdx.x, gy = blockIdx.y;
  const int lane = threadIdx.x;
  const int lm = lane & 15, hf = lane >> 4;
  const int colbase = gy * HD;

  v8f acc[5] = {};
  const f16* arow = A + (size_t)(rt * 16 + lm) * DMODEL;

  FragH fa0, fa1, fb0[5], fb1[5];

  fa0.q[0] = *(const uint4*)(arow + 8 * hf);
  fa0.q[1] = *(const uint4*)(arow + 16 + 8 * hf);
#pragma unroll
  for (int nt = 0; nt < 5; ++nt)
    load_bfrag(WprojT + (size_t)(colbase + nt * 16 + lm) * DMODEL + 8 * hf, fb0[nt]);

  for (int k0 = 0; k0 < DMODEL; k0 += 64) {
    fa1.q[0] = *(const uint4*)(arow + (k0 + 32) + 8 * hf);
    fa1.q[1] = *(const uint4*)(arow + (k0 + 32) + 16 + 8 * hf);
#pragma unroll
    for (int nt = 0; nt < 5; ++nt)
      load_bfrag(WprojT + (size_t)(colbase + nt * 16 + lm) * DMODEL + (k0 + 32) + 8 * hf,
                 fb1[nt]);
#pragma unroll
    for (int nt = 0; nt < 5; ++nt) acc[nt] = wmma16(fa0, fb0[nt], acc[nt]);

    if (k0 + 64 < DMODEL) {
      fa0.q[0] = *(const uint4*)(arow + (k0 + 64) + 8 * hf);
      fa0.q[1] = *(const uint4*)(arow + (k0 + 64) + 16 + 8 * hf);
#pragma unroll
      for (int nt = 0; nt < 5; ++nt)
        load_bfrag(WprojT + (size_t)(colbase + nt * 16 + lm) * DMODEL + (k0 + 64) + 8 * hf,
                   fb0[nt]);
    }
#pragma unroll
    for (int nt = 0; nt < 5; ++nt) acc[nt] = wmma16(fa1, fb1[nt], acc[nt]);
  }

#pragma unroll
  for (int nt = 0; nt < 5; ++nt)
#pragma unroll
    for (int v = 0; v < 8; ++v) {
      int row = rt * 16 + v + 8 * hf;
      int col = colbase + nt * 16 + lm;
      out[(size_t)row * DMODEL + col] = acc[nt][v] + bproj[col];
    }
}

// ---------------------------------------------------------------------------
extern "C" void kernel_launch(void* const* d_in, const int* in_sizes, int n_in,
                              void* d_out, int out_size, void* d_ws, size_t ws_size,
                              hipStream_t stream) {
  (void)in_sizes; (void)n_in; (void)out_size; (void)ws_size;
  const float* X     = (const float*)d_in[0];
  const float* cosp  = (const float*)d_in[1];
  const float* sinp  = (const float*)d_in[2];
  const float* wqkv  = (const float*)d_in[3];
  const float* bqkv  = (const float*)d_in[4];
  const float* wproj = (const float*)d_in[5];
  const float* bproj = (const float*)d_in[6];
  float* out = (float*)d_out;

  char* ws = (char*)d_ws;
  f16* WqkvT  = (f16*)ws; ws += (size_t)3 * DMODEL * DMODEL * sizeof(f16);
  f16* WprojT = (f16*)ws; ws += (size_t)DMODEL * DMODEL * sizeof(f16);
  f16* Qr     = (f16*)ws; ws += (size_t)CHUNKS * HEADS * CLEN * HDP * sizeof(f16);
  f16* Kr     = (f16*)ws; ws += (size_t)CHUNKS * HEADS * CLEN * HDP * sizeof(f16);
  f16* Vt     = (f16*)ws; ws += (size_t)CHUNKS * HEADS * HD * CLEN * sizeof(f16);
  f16* attnH  = (f16*)ws; ws += (size_t)TOTAL * DMODEL * sizeof(f16);

  {
    size_t tot = (size_t)3 * DMODEL * DMODEL;
    transpose_cvt<<<dim3((unsigned)((tot + 255) / 256)), 256, 0, stream>>>(
        wqkv, WqkvT, DMODEL, 3 * DMODEL);
  }
  {
    size_t tot = (size_t)DMODEL * DMODEL;
    transpose_cvt<<<dim3((unsigned)((tot + 255) / 256)), 256, 0, stream>>>(
        wproj, WprojT, DMODEL, DMODEL);
  }
  qkv_rope_kernel<<<dim3(TOTAL / 16, 48), 32, 0, stream>>>(
      X, WqkvT, bqkv, cosp, sinp, Qr, Kr, Vt);
  attn_kernel<<<dim3(CLEN / 16, HEADS, CHUNKS), 32, 0, stream>>>(Qr, Kr, Vt, attnH);
  proj_kernel<<<dim3(TOTAL / 16, DMODEL / HD), 32, 0, stream>>>(
      attnH, WprojT, bproj, out);
}
